// EuclideanCodebook_66606352827340
// MI455X (gfx1250) — matrix operand order repaired and checked by reading, not compiled, not run on hardware
//
#include <hip/hip_runtime.h>
#include <hip/hip_bf16.h>

// ---------------------------------------------------------------------------
// Problem constants (from reference): x[8,4096,256], embed[8192,256]
// ---------------------------------------------------------------------------
#define N_ROWS   32768      // 8*4096
#define K_CODES  8192
#define DIM      256
#define DECAY_F  0.8f
#define ONE_M_D  0.2f
#define EPS_F    1e-5f

#define WAVES_PER_BLOCK 16              // 512 threads
#define ROWS_PER_BLOCK  (WAVES_PER_BLOCK * 16)   // 256 x-rows per block
#define KD              32              // k-dim per WMMA
#define STEPS           (DIM / KD)      // 8 WMMAs per 16x16 tile
#define LDSROW          264             // padded row stride (halfwords): 528 B = 33*16
#define NTILES          (K_CODES / 16)  // 512 k-tiles

typedef __attribute__((ext_vector_type(16))) __bf16 v16bf;
typedef __attribute__((ext_vector_type(8)))  float  v8f;

union BfVec {
    v16bf v;
    uint4 q[2];
};

// fp32 -> bf16 bits, round-to-nearest-even
__device__ __forceinline__ unsigned short f2bf(float f) {
    unsigned int u = __float_as_uint(f);
    u += 0x7FFFu + ((u >> 16) & 1u);
    return (unsigned short)(u >> 16);
}

// ---------------------------------------------------------------------------
// Kernel: elementwise fp32 -> bf16 conversion
// ---------------------------------------------------------------------------
__global__ void k_f32_to_bf16(const float* __restrict__ in,
                              unsigned short* __restrict__ out, int n) {
    int i = blockIdx.x * blockDim.x + threadIdx.x;
    if (i < n) out[i] = f2bf(in[i]);
}

// ---------------------------------------------------------------------------
// Kernel: per-row squared norm of embed (one 256-thread block per row)
// ---------------------------------------------------------------------------
__global__ void k_row_norm(const float* __restrict__ e, float* __restrict__ enorm) {
    int k = blockIdx.x;
    float v = e[(size_t)k * DIM + threadIdx.x];
    float s = v * v;
    #pragma unroll
    for (int off = 16; off >= 1; off >>= 1) s += __shfl_xor(s, off, 32);
    __shared__ float wsum[8];
    if ((threadIdx.x & 31) == 0) wsum[threadIdx.x >> 5] = s;
    __syncthreads();
    if (threadIdx.x < 8) {
        float t = wsum[threadIdx.x];
        #pragma unroll
        for (int off = 4; off >= 1; off >>= 1) t += __shfl_xor(t, off, 32);
        if (threadIdx.x == 0) enorm[k] = t;
    }
}

// ---------------------------------------------------------------------------
// Kernel: zero a float region
// ---------------------------------------------------------------------------
__global__ void k_zero(float* __restrict__ p, int n) {
    int i = blockIdx.x * blockDim.x + threadIdx.x;
    if (i < n) p[i] = 0.0f;
}

// ---------------------------------------------------------------------------
// Main kernel: nearest-code argmax via bf16 WMMA with double-buffered
// async(global->LDS) staging of the 16-code B tile.
//
// Block = 512 threads = 16 waves; each wave owns one 16-row x-tile held in
// registers (64 VGPRs of A) and loops over all K codes in 16-wide tiles.
// Per iteration: wait own ASYNCcnt, one workgroup barrier, kick off the
// async fill of the *other* LDS buffer for tile t+1, then 8 WMMAs from the
// current buffer.  Score = 2*dot(x,e) - ||e||^2 (row-constant ||x||^2 drops).
// ---------------------------------------------------------------------------
__global__ __launch_bounds__(512)
void k_argmax_wmma(const unsigned short* __restrict__ xbf,   // [N, 256] bf16
                   const unsigned short* __restrict__ ebf,   // [K, 256] bf16
                   const float* __restrict__ enorm,          // [K]
                   int* __restrict__ bestIdx)                // [N]
{
    __shared__ unsigned short ldsB[2][16 * LDSROW];   // 2 x ~8.25 KB

    const int tid  = threadIdx.x;
    const int wave = tid >> 5;
    const int lane = tid & 31;
    const int half = lane >> 4;      // 0: low 16 lanes, 1: high 16 lanes
    const int lrow = lane & 15;

    const int m0 = blockIdx.x * ROWS_PER_BLOCK + wave * 16;

    // --- Preload this wave's A tile (16 rows x 256 d) into registers -------
    // ISA 16-bit A layout: lanes 0-15 hold K = {d0..d0+7}, lanes 16-31 hold
    // K = {d0+8..d0+15} in VGPR0-3; VGPR4-7 hold the +16 chunk.
    BfVec A[STEPS];
    const unsigned short* xrow = xbf + (size_t)(m0 + lrow) * DIM;
    #pragma unroll
    for (int s = 0; s < STEPS; ++s) {
        const int d0 = s * KD;
        A[s].q[0] = *(const uint4*)(xrow + d0 + half * 8);
        A[s].q[1] = *(const uint4*)(xrow + d0 + 16 + half * 8);
    }

    float best[8];
    int   bidx[8];
    #pragma unroll
    for (int v = 0; v < 8; ++v) { best[v] = -3.0e38f; bidx[v] = 0; }

    // --- async staging assignment: each thread owns one 16 B chunk ---------
    const int srow = tid >> 5;          // 0..15 (code row within tile)
    const int scol = (tid & 31) * 8;    // halfword offset 0..248

    // 32-bit LDS byte offsets (low 32 bits of the generic shared address)
    const unsigned lds0 = (unsigned)(size_t)&ldsB[0][srow * LDSROW + scol];
    const unsigned lds1 = (unsigned)(size_t)&ldsB[1][srow * LDSROW + scol];
    const unsigned short* gsrc = ebf + (size_t)srow * DIM + scol;

    // prime buffer 0 with tile 0 (ASYNCcnt-tracked, no VGPR round trip)
    asm volatile("global_load_async_to_lds_b128 %0, %1, off"
                 :: "v"(lds0), "v"(gsrc) : "memory");

    for (int t = 0; t < NTILES; ++t) {
        const int cur = t & 1;
        const int k0  = t * 16;

        asm volatile("s_wait_asynccnt 0x0" ::: "memory");
        __syncthreads();   // all waves' fills resident; prior buffer reads done

        if (t + 1 < NTILES) {
            const unsigned short* gnext = gsrc + (size_t)(t + 1) * 16 * DIM;
            asm volatile("global_load_async_to_lds_b128 %0, %1, off"
                         :: "v"(cur ? lds0 : lds1), "v"(gnext) : "memory");
        }

        v8f acc = {};
        const unsigned short* bbase =
            &ldsB[cur][lrow * LDSROW + half * 8];
        #pragma unroll
        for (int s = 0; s < STEPS; ++s) {
            // ISA 16-bit B layout (32x16): lane = output column (code),
            // lanes 0-15 hold K = {d0..d0+7, d0+16..d0+23}, lanes 16-31 hold {d0+8..d0+15, d0+24..d0+31}.
            BfVec B;
            B.q[0] = *(const uint4*)(bbase + s * KD);
            B.q[1] = *(const uint4*)(bbase + s * KD + 16);
            acc = __builtin_amdgcn_wmma_f32_16x16x32_bf16(
                false, A[s].v, false, B.v, (short)0, acc, false, false);
        }

        const float en  = enorm[k0 + lrow];
        const int   myk = k0 + lrow;
        #pragma unroll
        for (int v = 0; v < 8; ++v) {
            const float s = 2.0f * acc[v] - en;
            if (s > best[v]) { best[v] = s; bidx[v] = myk; }
        }
    }

    // --- Reduce over the 16 codes held across each 16-lane half ------------
    // C layout: VGPR v, lane l -> row M = v + 8*(l>>4), col N = l&15.
    #pragma unroll
    for (int v = 0; v < 8; ++v) {
        float bv = best[v];
        int   bi = bidx[v];
        #pragma unroll
        for (int off = 8; off >= 1; off >>= 1) {
            float ov = __shfl_xor(bv, off, 32);
            int   oi = __shfl_xor(bi, off, 32);
            if (ov > bv || (ov == bv && oi < bi)) { bv = ov; bi = oi; }
        }
        if (lrow == 0) bestIdx[m0 + v + 8 * half] = bi;
    }
}

// ---------------------------------------------------------------------------
// Kernel: gather quantize rows, emit indices (as float), scatter EMA stats.
// One 256-thread block per x-row.
// ---------------------------------------------------------------------------
__global__ void k_gather_scatter(const float* __restrict__ x,
                                 const float* __restrict__ embed,
                                 const int* __restrict__ idx,
                                 float* __restrict__ quant,
                                 float* __restrict__ indOut,
                                 float* __restrict__ counts,
                                 float* __restrict__ esum)
{
    const int row = blockIdx.x;
    const int k   = idx[row];
    const int d   = threadIdx.x;
    const float xv = x[(size_t)row * DIM + d];
    quant[(size_t)row * DIM + d] = embed[(size_t)k * DIM + d];
    atomicAdd(&esum[(size_t)k * DIM + d], xv);
    if (d == 0) {
        indOut[row] = (float)k;
        atomicAdd(&counts[k], 1.0f);
    }
}

// ---------------------------------------------------------------------------
// Kernel: new_cluster_size EMA + global sum (atomic per wave)
// ---------------------------------------------------------------------------
__global__ void k_ema_counts(const float* __restrict__ cs,
                             const float* __restrict__ counts,
                             float* __restrict__ ncs,
                             float* __restrict__ total)
{
    const int k = blockIdx.x * blockDim.x + threadIdx.x;   // K divisible by 256
    float v = cs[k] * DECAY_F + counts[k] * ONE_M_D;
    ncs[k] = v;
    #pragma unroll
    for (int off = 16; off >= 1; off >>= 1) v += __shfl_xor(v, off, 32);
    if ((threadIdx.x & 31) == 0) atomicAdd(total, v);
}

// ---------------------------------------------------------------------------
// Kernel: new_embed_avg EMA + Laplace-smoothed normalization
// ---------------------------------------------------------------------------
__global__ void k_ema_embed(const float* __restrict__ eavg,
                            const float* __restrict__ esum,
                            const float* __restrict__ ncs,
                            const float* __restrict__ total,
                            float* __restrict__ navg,
                            float* __restrict__ nembed)
{
    const int i = blockIdx.x * blockDim.x + threadIdx.x;   // K*DIM
    const int k = i >> 8;
    const float a = eavg[i] * DECAY_F + esum[i] * ONE_M_D;
    navg[i] = a;
    const float t  = *total;
    const float sm = (ncs[k] + EPS_F) / (t + (float)K_CODES * EPS_F) * t;
    nembed[i] = a / sm;
}

// ---------------------------------------------------------------------------
// Workspace layout (bytes).  Requires ~30 MB of scratch.
// ---------------------------------------------------------------------------
static const size_t OFF_XBF   = 0;                       // N*DIM bf16   = 16 MB
static const size_t OFF_EBF   = 16777216;                // K*DIM bf16   =  4 MB
static const size_t OFF_ENORM = 20971520;                // K f32        = 32 KB
static const size_t OFF_IDX   = 21004288;                // N i32        = 128 KB
static const size_t OFF_CNT   = 21135360;                // K f32        = 32 KB
static const size_t OFF_ESUM  = 21168128;                // K*DIM f32    =  8 MB
static const size_t OFF_TOT   = 29556736;                // 1 f32

// Output layout (floats, concatenated in reference return order)
static const size_t OUT_Q    = 0;          // quantize        N*DIM
static const size_t OUT_IND  = 8388608;    // embed_ind       N
static const size_t OUT_NCS  = 8421376;    // new_cluster_sz  K
static const size_t OUT_NAVG = 8429568;    // new_embed_avg   K*DIM
static const size_t OUT_NEMB = 10526720;   // new_embed       K*DIM

extern "C" void kernel_launch(void* const* d_in, const int* in_sizes, int n_in,
                              void* d_out, int out_size, void* d_ws, size_t ws_size,
                              hipStream_t stream) {
    const float* x     = (const float*)d_in[0];   // [N, DIM]
    const float* embed = (const float*)d_in[1];   // [K, DIM]
    const float* cs    = (const float*)d_in[2];   // [K]
    const float* eavg  = (const float*)d_in[3];   // [K, DIM]

    char* ws = (char*)d_ws;
    unsigned short* xbf   = (unsigned short*)(ws + OFF_XBF);
    unsigned short* ebf   = (unsigned short*)(ws + OFF_EBF);
    float*          enorm = (float*)(ws + OFF_ENORM);
    int*            idx   = (int*)(ws + OFF_IDX);
    float*          cnt   = (float*)(ws + OFF_CNT);
    float*          esum  = (float*)(ws + OFF_ESUM);
    float*          total = (float*)(ws + OFF_TOT);

    float* out    = (float*)d_out;
    float* quant  = out + OUT_Q;
    float* indOut = out + OUT_IND;
    float* ncs    = out + OUT_NCS;
    float* navg   = out + OUT_NAVG;
    float* nembed = out + OUT_NEMB;

    // 1) convert x and embed to bf16
    {
        int n = N_ROWS * DIM;
        k_f32_to_bf16<<<(n + 255) / 256, 256, 0, stream>>>(x, xbf, n);
    }
    {
        int n = K_CODES * DIM;
        k_f32_to_bf16<<<(n + 255) / 256, 256, 0, stream>>>(embed, ebf, n);
    }

    // 2) ||e||^2 per code
    k_row_norm<<<K_CODES, 256, 0, stream>>>(embed, enorm);

    // 3) zero scatter buffers (counts + esum + total are contiguous)
    {
        int n = K_CODES + K_CODES * DIM + 1;
        k_zero<<<(n + 255) / 256, 256, 0, stream>>>(cnt, n);
    }

    // 4) WMMA argmax: 256 rows per block, async double-buffered B staging
    k_argmax_wmma<<<N_ROWS / ROWS_PER_BLOCK, WAVES_PER_BLOCK * 32, 0, stream>>>(
        xbf, ebf, enorm, idx);

    // 5) gather quantize + indices, scatter counts / embed_sum
    k_gather_scatter<<<N_ROWS, DIM, 0, stream>>>(x, embed, idx, quant, indOut, cnt, esum);

    // 6) cluster-size EMA + total
    k_ema_counts<<<K_CODES / 256, 256, 0, stream>>>(cs, cnt, ncs, total);

    // 7) embed_avg EMA + normalized new_embed
    k_ema_embed<<<(K_CODES * DIM) / 256, 256, 0, stream>>>(eavg, esum, ncs, total,
                                                           navg, nembed);
}
